// DivergenceRegularizer_31233002177072
// MI455X (gfx1250) — matrix operand order repaired and checked by reading, not compiled, not run on hardware
//
#include <hip/hip_runtime.h>

#define N_NODES 4096
#define D_FEAT  256
#define BATCH   8

#define TILE_M 128
#define TILE_N 128
#define TILE_K 64
#define LDSROW 72                    // padded row (f16 elems): 144B stride, 16B-aligned
#define LDS_TILE (TILE_M * LDSROW)   // 9216 f16 = 18432 B per matrix tile

typedef _Float16 half8 __attribute__((ext_vector_type(8)));
typedef _Float16 v16h  __attribute__((ext_vector_type(16)));
typedef float    v8f   __attribute__((ext_vector_type(8)));

// ---------------------------------------------------------------------------
// Kernel 1: adjacency (int32, {0,1}) -> f16 A matrix + float invDeg[row]
// (invDeg = 1/deg for deg>0 else 0; doubles as the isolated-node mask)
// ---------------------------------------------------------------------------
__global__ __launch_bounds__(256) void prep_adj(const int* __restrict__ adj,
                                                _Float16* __restrict__ A,
                                                float* __restrict__ invDeg) {
    __shared__ int red[256];
    const int row = blockIdx.x;
    const int* arow = adj + (size_t)row * N_NODES;
    _Float16* Arow  = A   + (size_t)row * N_NODES;
    int cnt = 0;
    for (int j = threadIdx.x; j < N_NODES; j += 256) {
        int nz = (arow[j] != 0);
        cnt += nz;
        Arow[j] = (_Float16)(float)nz;
    }
    red[threadIdx.x] = cnt;
    __syncthreads();
    for (int s = 128; s > 0; s >>= 1) {
        if (threadIdx.x < s) red[threadIdx.x] += red[threadIdx.x + s];
        __syncthreads();
    }
    if (threadIdx.x == 0)
        invDeg[row] = (red[0] > 0) ? (1.0f / (float)red[0]) : 0.0f;
}

// ---------------------------------------------------------------------------
// Kernel 2: S (B,N,d) f32 -> transposed f16 hi/lo:  BT[c][j], c = b*256+k
// ---------------------------------------------------------------------------
__global__ __launch_bounds__(256) void prep_S(const float* __restrict__ S,
                                              _Float16* __restrict__ BhiT,
                                              _Float16* __restrict__ BloT) {
    const int idx = blockIdx.x * 256 + threadIdx.x;     // < 2^23
    float s = S[idx];
    _Float16 hi = (_Float16)s;
    _Float16 lo = (_Float16)(s - (float)hi);
    int k = idx & (D_FEAT - 1);
    int j = (idx >> 8) & (N_NODES - 1);
    int b = idx >> 20;                                  // / (4096*256)
    size_t o = (size_t)(b * D_FEAT + k) * N_NODES + j;
    BhiT[o] = hi;
    BloT[o] = lo;
}

// ---------------------------------------------------------------------------
// Kernel 3: WMMA GEMM, async global->LDS double buffering (TILE_K=64 slabs
// so each prefetch is covered by 32 WMMAs/wave of compute), fused divergence
// epilogue. 8 waves, 128x128 block tile, 64x32 wave tile.
// ---------------------------------------------------------------------------
__global__ __launch_bounds__(256) void gemm_div(const _Float16* __restrict__ A,
                                                const _Float16* __restrict__ BhiT,
                                                const _Float16* __restrict__ BloT,
                                                const float* __restrict__ invDeg,
                                                const float* __restrict__ S,
                                                float* __restrict__ partials) {
    // [buf 0|1] x [A, Bhi, Blo] tiles
    __shared__ __align__(16) _Float16 smem[6 * LDS_TILE];
    __shared__ float sred[256];

    const int tid   = threadIdx.x;
    const int lane  = tid & 31;
    const int wave  = tid >> 5;
    const int waveM = wave & 1;      // 2 waves x 64 rows
    const int waveN = wave >> 1;     // 4 waves x 32 cols

    const int bm = blockIdx.x & 31;
    const int bn = blockIdx.x >> 5;
    const int rowBase = bm * TILE_M;
    const int colBase = bn * TILE_N;

    // Low 32 bits of the generic shared pointer = LDS byte offset.
    const unsigned ldsBase = (unsigned)(size_t)smem;

    v8f acc[4][2];
    {
        v8f z = {0.f, 0.f, 0.f, 0.f, 0.f, 0.f, 0.f, 0.f};
        for (int i = 0; i < 4; i++)
            for (int j = 0; j < 2; j++) acc[i][j] = z;
    }

    // Stage one 128x64 K-slab of all three matrices into LDS buffer `buf`
    // via async global->LDS. 1024 16B chunks per matrix; chunk id = t*256+tid
    // so every instruction is lane-contiguous (fully coalesced).
    auto stage = [&](int buf, int k0) {
        const unsigned base = ldsBase + (unsigned)(buf * 3 * LDS_TILE * 2);
        for (int t = 0; t < 4; t++) {
            int cid = t * 256 + tid;       // 0..1023
            int r   = cid >> 3;
            int off = (cid & 7) * 8;
            unsigned le = (unsigned)((r * LDSROW + off) * 2);
            const void* ga = (const void*)(A    + (size_t)(rowBase + r) * N_NODES + k0 + off);
            const void* gh = (const void*)(BhiT + (size_t)(colBase + r) * N_NODES + k0 + off);
            const void* gl = (const void*)(BloT + (size_t)(colBase + r) * N_NODES + k0 + off);
            asm volatile("global_load_async_to_lds_b128 %0, %1, off"
                         :: "v"(base + le), "v"(ga) : "memory");
            asm volatile("global_load_async_to_lds_b128 %0, %1, off"
                         :: "v"(base + (unsigned)(2 * LDS_TILE) + le), "v"(gh) : "memory");
            asm volatile("global_load_async_to_lds_b128 %0, %1, off"
                         :: "v"(base + (unsigned)(4 * LDS_TILE) + le), "v"(gl) : "memory");
        }
    };

    const int mrow = lane & 15;
    const int kbA  = (lane >> 4) * 8;    // ISA A-frag: lanes>=16 hold K+8
    const int kbB  = (lane >> 4) * 8;    // ISA B-frag: lanes>=16 hold K+8
    const int NSTEP = N_NODES / TILE_K;  // 64

    // Prologue: fill buffer 0 with K-slab 0.
    stage(0, 0);
    asm volatile("s_wait_asynccnt 0" ::: "memory");
    __syncthreads();

    for (int ks = 0; ks < NSTEP; ks++) {
        const int p = ks & 1;
        if (ks + 1 < NSTEP) stage(p ^ 1, (ks + 1) * TILE_K);

        const _Float16* cA  = smem + (p * 3 + 0) * LDS_TILE;
        const _Float16* cBh = smem + (p * 3 + 1) * LDS_TILE;
        const _Float16* cBl = smem + (p * 3 + 2) * LDS_TILE;

        // Two 32-wide sub-steps within the staged 64-wide slab.
        for (int kk = 0; kk < TILE_K; kk += 32) {
            union Frag { v16h v; half8 h[2]; };
            Frag af[4], bh[2], bl[2];
            for (int tm = 0; tm < 4; tm++) {
                int r = waveM * 64 + tm * 16 + mrow;
                af[tm].h[0] = *(const half8*)(cA + r * LDSROW + kk + kbA);
                af[tm].h[1] = *(const half8*)(cA + r * LDSROW + kk + 16 + kbA);
            }
            for (int tn = 0; tn < 2; tn++) {
                int c = waveN * 32 + tn * 16 + mrow;
                bh[tn].h[0] = *(const half8*)(cBh + c * LDSROW + kk + kbB);
                bh[tn].h[1] = *(const half8*)(cBh + c * LDSROW + kk + kbB + 16);
                bl[tn].h[0] = *(const half8*)(cBl + c * LDSROW + kk + kbB);
                bl[tn].h[1] = *(const half8*)(cBl + c * LDSROW + kk + kbB + 16);
            }
            for (int tm = 0; tm < 4; tm++)
                for (int tn = 0; tn < 2; tn++) {
                    acc[tm][tn] = __builtin_amdgcn_wmma_f32_16x16x32_f16(
                        false, af[tm].v, false, bh[tn].v, (short)0, acc[tm][tn], false, false);
                    acc[tm][tn] = __builtin_amdgcn_wmma_f32_16x16x32_f16(
                        false, af[tm].v, false, bl[tn].v, (short)0, acc[tm][tn], false, false);
                }
        }

        // My async fills of buf p^1 must land before anyone reads them.
        asm volatile("s_wait_asynccnt 0" ::: "memory");
        __syncthreads();
    }

    // Fused epilogue: (C*invDeg - S)^2, zero for isolated nodes (invDeg==0).
    float lsum = 0.0f;
    for (int tm = 0; tm < 4; tm++) {
        for (int tn = 0; tn < 2; tn++) {
            int c = colBase + waveN * 32 + tn * 16 + (lane & 15);
            int b = c >> 8;
            int k = c & 255;
            for (int v = 0; v < 8; v++) {
                int i = rowBase + waveM * 64 + tm * 16 + v + ((lane >> 4) * 8);
                float inv = invDeg[i];
                float m   = (inv != 0.0f) ? 1.0f : 0.0f;
                float val = acc[tm][tn][v] * inv
                          - S[((size_t)b * N_NODES + i) * D_FEAT + k];
                lsum += m * val * val;
            }
        }
    }
    sred[tid] = lsum;
    __syncthreads();
    for (int s = 128; s > 0; s >>= 1) {
        if (tid < s) sred[tid] += sred[tid + s];
        __syncthreads();
    }
    if (tid == 0) partials[blockIdx.x] = sred[0];
}

// ---------------------------------------------------------------------------
// Kernel 4: deterministic final reduction of 512 partials -> scalar.
// ---------------------------------------------------------------------------
__global__ __launch_bounds__(256) void reduce_out(const float* __restrict__ partials,
                                                  float* __restrict__ out) {
    __shared__ float sred[256];
    sred[threadIdx.x] = partials[threadIdx.x] + partials[threadIdx.x + 256];
    __syncthreads();
    for (int s = 128; s > 0; s >>= 1) {
        if (threadIdx.x < s) sred[threadIdx.x] += sred[threadIdx.x + s];
        __syncthreads();
    }
    if (threadIdx.x == 0)
        out[0] = sred[0] / (float)((size_t)BATCH * N_NODES * D_FEAT);
}

// ---------------------------------------------------------------------------
extern "C" void kernel_launch(void* const* d_in, const int* in_sizes, int n_in,
                              void* d_out, int out_size, void* d_ws, size_t ws_size,
                              hipStream_t stream) {
    const float* S   = (const float*)d_in[0];   // (8, 4096, 256) f32
    const int*   adj = (const int*)d_in[1];     // (4096, 4096) int32

    char* ws = (char*)d_ws;
    _Float16* A     = (_Float16*)(ws);                          // 32 MB
    _Float16* BhiT  = (_Float16*)(ws + (size_t)33554432);       // 16 MB
    _Float16* BloT  = (_Float16*)(ws + (size_t)50331648);       // 16 MB
    float*    invD  = (float*)   (ws + (size_t)67108864);       // 16 KB
    float*    parts = (float*)   (ws + (size_t)67125248);       // 2 KB
    float*    out   = (float*)d_out;

    hipLaunchKernelGGL(prep_adj, dim3(N_NODES), dim3(256), 0, stream, adj, A, invD);
    hipLaunchKernelGGL(prep_S, dim3((BATCH * N_NODES * D_FEAT) / 256), dim3(256), 0, stream,
                       S, BhiT, BloT);
    hipLaunchKernelGGL(gemm_div, dim3(512), dim3(256), 0, stream,
                       A, BhiT, BloT, invD, S, parts);
    hipLaunchKernelGGL(reduce_out, dim3(1), dim3(256), 0, stream, parts, out);
}
